// MambaLayer_91302414779055
// MI455X (gfx1250) — matrix operand order repaired and checked by reading, not compiled, not run on hardware
//
#include <hip/hip_runtime.h>

typedef __attribute__((ext_vector_type(16))) __bf16 v16bf;
typedef __attribute__((ext_vector_type(8)))  float  v8f;

#define BB     4
#define LSEQ   4096
#define DMODEL 128
#define DINNER 256
#define DSTATE 16
#define ROWS   (BB * LSEQ)   // 16384

// ---------------------------------------------------------------------------
// Transpose-convert fp32 weight W[K][N] (row-major) -> bf16 Wt[N][K]
// ---------------------------------------------------------------------------
__global__ void cvt_transpose_bf16(const float* __restrict__ W,
                                   __bf16* __restrict__ Wt, int K, int N) {
    int i = blockIdx.x * blockDim.x + threadIdx.x;
    if (i >= K * N) return;
    int n = i / K;
    int k = i - n * K;
    Wt[i] = (__bf16)W[k * N + n];
}

// ---------------------------------------------------------------------------
// bf16-WMMA GEMM, all dims compile-time:
//   C[M][LDC] = act(A[M][LDA](fp32) @ Wt[N][K](bf16) + bias)
// Each wave computes a 16 x (16*NT) strip: one A fragment per K-step feeds
// NT v_wmma_f32_16x16x32_bf16 into NT accumulators (A reuse = NT).
// K loop fully unrolled; all strides fold into immediate offsets.
// 8 waves (256 threads) per block. ACT: 0 = identity, 1 = softplus.
// ---------------------------------------------------------------------------
template <int ACT, int NT, int LDA, int LDC, int N, int K, bool HASBIAS>
__global__ void gemm_wmma_bf16(const float* __restrict__ A,
                               const __bf16* __restrict__ Wt,
                               const float* __restrict__ bias,
                               float* __restrict__ C) {
    constexpr int groupsN = (N >> 4) / NT;   // tile-groups along N

    const int lane = threadIdx.x & 31;
    const int wave = threadIdx.x >> 5;

    const int group = blockIdx.x * 8 + wave;
    const int tn0 = (group % groupsN) * NT;  // first N-tile of this wave
    const int tm  = group / groupsN;         // M-tile

    const int g  = lane >> 4;                // lane group 0/1
    const int mn = lane & 15;                // A row within tile / C column within tile

    const int rowA = tm * 16 + mn;

    v8f acc[NT];
    const __bf16* Bcol[NT];
#pragma unroll
    for (int j = 0; j < NT; ++j) {
        const int coln = (tn0 + j) * 16 + mn;
        const float bv = HASBIAS ? bias[coln] : 0.0f;
#pragma unroll
        for (int r = 0; r < 8; ++r) acc[j][r] = bv;
        Bcol[j] = Wt + (long)coln * K;
    }

    const float* Arow = A + (long)rowA * LDA;

#pragma unroll
    for (int kt = 0; kt < K; kt += 32) {
        // ---- A fragment: lane covers K = kt+g*8 .. +7 and kt+16+g*8 .. +7
        const int k0 = kt + g * 8;
        float4 f0 = *(const float4*)(Arow + k0);
        float4 f1 = *(const float4*)(Arow + k0 + 4);
        float4 f2 = *(const float4*)(Arow + k0 + 16);
        float4 f3 = *(const float4*)(Arow + k0 + 20);
        v16bf a;
        a[0]  = (__bf16)f0.x; a[1]  = (__bf16)f0.y; a[2]  = (__bf16)f0.z; a[3]  = (__bf16)f0.w;
        a[4]  = (__bf16)f1.x; a[5]  = (__bf16)f1.y; a[6]  = (__bf16)f1.z; a[7]  = (__bf16)f1.w;
        a[8]  = (__bf16)f2.x; a[9]  = (__bf16)f2.y; a[10] = (__bf16)f2.z; a[11] = (__bf16)f2.w;
        a[12] = (__bf16)f3.x; a[13] = (__bf16)f3.y; a[14] = (__bf16)f3.z; a[15] = (__bf16)f3.w;

        // ---- NT B fragments: each lane covers K = kt+g*8 .. +7 and kt+16+g*8 .. +7
#pragma unroll
        for (int j = 0; j < NT; ++j) {
            union { uint4 u[2]; v16bf v; } ub;
            const uint4* bp = (const uint4*)(Bcol[j] + kt + g * 8);
            ub.u[0] = bp[0];
            ub.u[1] = bp[2];
            acc[j] = __builtin_amdgcn_wmma_f32_16x16x32_bf16(
                /*neg_a=*/false, a, /*neg_b=*/false, ub.v,
                /*c_mod=*/(short)0, acc[j], /*reuse_a=*/false, /*reuse_b=*/false);
        }
    }

    // Epilogue: one base address per (lane, j); 8 stores at immediate offsets.
    const int row0 = tm * 16 + 8 * g;        // first C row for this lane
#pragma unroll
    for (int j = 0; j < NT; ++j) {
        const int coln = (tn0 + j) * 16 + mn;
        float* Cbase = C + (long)row0 * LDC + coln;
#pragma unroll
        for (int r = 0; r < 8; ++r) {
            float v = acc[j][r];
            if (ACT == 1) {  // softplus
                v = (v > 20.0f) ? v : __logf(1.0f + __expf(v));
            }
            Cbase[r * LDC] = v;
        }
    }
}

// ---------------------------------------------------------------------------
// Depthwise causal 4-tap conv over L + bias + SiLU.
// x_in = xz[:, 0:256]; one thread per (row, channel).
// ---------------------------------------------------------------------------
__global__ void conv_silu_kernel(const float* __restrict__ xz,
                                 const float* __restrict__ conv_w,
                                 const float* __restrict__ conv_b,
                                 float* __restrict__ xc) {
    const int r = blockIdx.x;        // 0..16383
    const int d = threadIdx.x;       // 0..255
    const int t = r & (LSEQ - 1);

    const float w0 = conv_w[d * 4 + 0];
    const float w1 = conv_w[d * 4 + 1];
    const float w2 = conv_w[d * 4 + 2];
    const float w3 = conv_w[d * 4 + 3];

    const float* base = xz + (long)r * (2 * DINNER) + d;
    float acc = conv_b[d] + w3 * base[0];
    if (t >= 1) acc += w2 * base[-(long)(2 * DINNER)];
    if (t >= 2) acc += w1 * base[-(long)(4 * DINNER)];
    if (t >= 3) acc += w0 * base[-(long)(6 * DINNER)];

    acc = acc / (1.0f + __expf(-acc));  // SiLU
    xc[(long)r * DINNER + d] = acc;
}

// ---------------------------------------------------------------------------
// Selective scan. One block per batch, one thread per channel d.
// h[16] kept in VGPRs; per-step B/C (32 floats) staged through LDS
// (double-buffered, single barrier per step). Fused +u*D and *silu(z).
// yf may alias xc (each element is read before being overwritten).
// ---------------------------------------------------------------------------
__global__ void scan_kernel(const float* __restrict__ xc,
                            const float* __restrict__ dtb,
                            const float* __restrict__ xdbl,
                            const float* __restrict__ xz,
                            const float* __restrict__ A_log,
                            const float* __restrict__ Dp,
                            float* __restrict__ yf) {
    __shared__ float shBC[2][32];
    const int b = blockIdx.x;
    const int d = threadIdx.x;

    float An[DSTATE];
#pragma unroll
    for (int n = 0; n < DSTATE; ++n) An[n] = -__expf(A_log[d * DSTATE + n]);
    const float Dd = Dp[d];

    float h[DSTATE];
#pragma unroll
    for (int n = 0; n < DSTATE; ++n) h[n] = 0.0f;

    for (int t = 0; t < LSEQ; ++t) {
        const long r = (long)b * LSEQ + t;
        if (d < 32) shBC[t & 1][d] = xdbl[r * 288 + DINNER + d];
        __syncthreads();

        const float u   = xc[r * DINNER + d];
        const float dtv = dtb[r * DINNER + d];
        const float zv  = xz[r * (2 * DINNER) + DINNER + d];
        const float du  = dtv * u;

        const float* Bt = shBC[t & 1];
        const float* Ct = shBC[t & 1] + DSTATE;

        float y = 0.0f;
#pragma unroll
        for (int n = 0; n < DSTATE; ++n) {
            const float a = __expf(dtv * An[n]);
            h[n] = a * h[n] + du * Bt[n];
            y += h[n] * Ct[n];
        }

        const float sz = zv / (1.0f + __expf(-zv));
        yf[r * DINNER + d] = (y + u * Dd) * sz;
    }
}

// ---------------------------------------------------------------------------
// Launcher
// ---------------------------------------------------------------------------
extern "C" void kernel_launch(void* const* d_in, const int* in_sizes, int n_in,
                              void* d_out, int out_size, void* d_ws, size_t ws_size,
                              hipStream_t stream) {
    const float* x      = (const float*)d_in[0];
    const float* W_in   = (const float*)d_in[1];
    const float* b_in   = (const float*)d_in[2];
    const float* conv_w = (const float*)d_in[3];
    const float* conv_b = (const float*)d_in[4];
    const float* W_x    = (const float*)d_in[5];
    const float* W_dt   = (const float*)d_in[6];
    const float* b_dt   = (const float*)d_in[7];
    const float* A_log  = (const float*)d_in[8];
    const float* Dvec   = (const float*)d_in[9];
    const float* W_out  = (const float*)d_in[10];
    const float* b_out  = (const float*)d_in[11];
    float* out = (float*)d_out;

    char* ws = (char*)d_ws;
    size_t off = 0;
    float* xz   = (float*)(ws + off); off += (size_t)ROWS * 512 * 4;  // 32 MB
    float* xc   = (float*)(ws + off); off += (size_t)ROWS * 256 * 4;  // 16 MB (also yf)
    float* xdbl = (float*)(ws + off); off += (size_t)ROWS * 288 * 4;  // 18 MB
    float* dtb  = (float*)(ws + off); off += (size_t)ROWS * 256 * 4;  // 16 MB
    __bf16* wtin  = (__bf16*)(ws + off); off += (size_t)512 * 128 * 2;
    __bf16* wtx   = (__bf16*)(ws + off); off += (size_t)288 * 256 * 2;
    __bf16* wtdt  = (__bf16*)(ws + off); off += (size_t)256 * 256 * 2;
    __bf16* wtout = (__bf16*)(ws + off); off += (size_t)128 * 256 * 2;
    (void)ws_size; (void)in_sizes; (void)n_in; (void)out_size;

    // 1) weight conversion (fp32 [K][N] -> bf16 [N][K])
    cvt_transpose_bf16<<<(512 * 128 + 255) / 256, 256, 0, stream>>>(W_in,  wtin,  128, 512);
    cvt_transpose_bf16<<<(288 * 256 + 255) / 256, 256, 0, stream>>>(W_x,   wtx,   256, 288);
    cvt_transpose_bf16<<<(256 * 256 + 255) / 256, 256, 0, stream>>>(W_dt,  wtdt,  256, 256);
    cvt_transpose_bf16<<<(128 * 256 + 255) / 256, 256, 0, stream>>>(W_out, wtout, 256, 128);

    // 2) xz = x @ W_in + b_in        [16384 x 512], K=128, NT=4
    //    tile-groups = (16384/16) * (512/64) = 8192 -> 1024 blocks of 8 waves
    gemm_wmma_bf16<0, 4, 128, 512, 512, 128, true>
        <<<(ROWS / 16) * (512 / 64) / 8, 256, 0, stream>>>(x, wtin, b_in, xz);

    // 3) depthwise causal conv + SiLU -> xc  [16384 x 256]
    conv_silu_kernel<<<ROWS, 256, 0, stream>>>(xz, conv_w, conv_b, xc);

    // 4) x_dbl = xc @ W_x            [16384 x 288], K=256, NT=3 (18 tiles / 3)
    //    tile-groups = 1024 * 6 = 6144 -> 768 blocks
    gemm_wmma_bf16<0, 3, 256, 288, 288, 256, false>
        <<<(ROWS / 16) * ((288 / 16) / 3) / 8, 256, 0, stream>>>(xc, wtx, nullptr, xdbl);

    // 5) dt = softplus(dt_in @ W_dt + b_dt)  [16384 x 256], K=256, NT=4
    //    tile-groups = 1024 * 4 = 4096 -> 512 blocks
    gemm_wmma_bf16<1, 4, 288, 256, 256, 256, true>
        <<<(ROWS / 16) * (256 / 64) / 8, 256, 0, stream>>>(xdbl, wtdt, b_dt, dtb);

    // 6) selective scan (y fused with +u*D and *silu(z)); output aliases xc
    scan_kernel<<<BB, DINNER, 0, stream>>>(xc, dtb, xdbl, xz, A_log, Dvec, xc);

    // 7) out = y @ W_out + b_out     [16384 x 128], K=256, NT=4
    //    tile-groups = 1024 * 2 = 2048 -> 256 blocks
    gemm_wmma_bf16<0, 4, 256, 128, 128, 256, true>
        <<<(ROWS / 16) * (128 / 64) / 8, 256, 0, stream>>>(xc, wtout, b_out, out);
}
